// VQVAE_22308060135451
// MI455X (gfx1250) — matrix operand order repaired and checked by reading, not compiled, not run on hardware
//
#include <hip/hip_runtime.h>
#include <stdint.h>

// VQ-VAE quantization for MI455X (gfx1250), wave32 + WMMA bf16.
// N=32768 rows, D=256, K=1024 codes.
// argmin(k) of ||z-e_k||^2  ==  argmin(k) of ||e_k||^2 - 2 z.e_k   (||z||^2 constant per row)

#define N_ROWS 32768
#define DIM    256
#define KCODES 1024
#define MBLK   64      // rows per workgroup in fused kernel
#define APAD   264     // 256 + 8 halves padding (bank spread, keeps 16B align: 528B rows)

typedef __attribute__((ext_vector_type(16))) __bf16 v16bf;
typedef __attribute__((ext_vector_type(8)))  float  v8f;

union Frag {
    v16bf v;
    uint4 q[2];
};

// round-to-nearest-even f32 -> bf16 (bit pattern)
__device__ __forceinline__ unsigned short f2bf(float x) {
    union { float f; unsigned u; } c;
    c.f = x;
    unsigned u = c.u + 0x7FFFu + ((c.u >> 16) & 1u);
    return (unsigned short)(u >> 16);
}

// ---------------------------------------------------------------------------
// Kernel 0: exact f32 codebook norms ||e_k||^2 + one-time bf16 codebook copy.
// ---------------------------------------------------------------------------
__global__ __launch_bounds__(256) void cbprep_kernel(const float* __restrict__ cbk,
                                                     float* __restrict__ norms,
                                                     unsigned short* __restrict__ bfcb) {
    __shared__ float red[256];
    const int code = blockIdx.x;
    const float v = cbk[(size_t)code * DIM + threadIdx.x];
    bfcb[(size_t)code * DIM + threadIdx.x] = f2bf(v);
    red[threadIdx.x] = v * v;
    __syncthreads();
    for (int s = 128; s > 0; s >>= 1) {
        if (threadIdx.x < s) red[threadIdx.x] += red[threadIdx.x + s];
        __syncthreads();
    }
    if (threadIdx.x == 0) norms[code] = red[0];
}

// ---------------------------------------------------------------------------
// Kernel 1 (fused): WMMA scores + argmin + gather + outputs + loss partials.
// Block = 256 threads = 8 wave32s; MBLK=64 rows vs all 1024 codes.
// Wave w: row-tile rt = w&3, column half ch = w>>2.
// A fragments hoisted into registers (loop-invariant over codebook chunks).
// Next B chunk brought into WGP$ via global_prefetch during the WMMA loop
// (zero register cost), then copied LDS-ward synchronously at chunk top.
// B fragments double-buffered in registers for ds/wmma overlap.
// ---------------------------------------------------------------------------
__global__ __launch_bounds__(256, 1) void vq_fused_kernel(const float* __restrict__ z,
                                                          const float* __restrict__ cbk,
                                                          const float* __restrict__ norms,
                                                          const unsigned short* __restrict__ bfcb,
                                                          float* __restrict__ outZ,
                                                          float* __restrict__ outQ,
                                                          float* __restrict__ partial) {
    __shared__ __align__(16) unsigned short Abf[MBLK][APAD]; // 33792 B
    __shared__ __align__(16) unsigned short Bbf[32][APAD];   // 16896 B
    __shared__ float cval[2][MBLK];
    __shared__ int   cind[2][MBLK];
    __shared__ int   finalIdx[MBLK];
    __shared__ float red[256];

    const int tid     = threadIdx.x;
    const int rowBase = blockIdx.x * MBLK;

    // Stage A: 64 rows of z as bf16 (coalesced global reads; 64 KB per block)
    for (int i = tid; i < MBLK * DIM; i += 256) {
        const int r = i >> 8, c = i & 255;
        Abf[r][c] = f2bf(z[(size_t)(rowBase + r) * DIM + c]);
    }

    const int lane = tid & 31;
    const int w    = tid >> 5;
    const int hl   = lane >> 4;   // half-wave
    const int m    = lane & 15;
    const int rt   = w & 3;       // row tile
    const int ch   = w >> 2;      // column half
    const int rowA      = rt * 16 + m;
    const int codeLocal = ch * 16 + m;

    __syncthreads();

    // Hoist all A fragments for this wave's row tile into registers:
    // 8 k-steps x 32B = 64 VGPRs, loop-invariant across all codebook chunks.
    Frag aF[8];
#pragma unroll
    for (int ks = 0; ks < 8; ks++) {
        const int kb = ks * 32;
        // A k order used here (contiguous, same as B): lanes 0-15 K=kb..kb+15; lanes 16-31 K=kb+16..kb+31
        aF[ks].q[0] = *reinterpret_cast<const uint4*>(&Abf[rowA][kb + hl * 16]);
        aF[ks].q[1] = *reinterpret_cast<const uint4*>(&Abf[rowA][kb + hl * 16 + 8]);
    }

    float minv[8];
    int   mini[8];
#pragma unroll
    for (int j = 0; j < 8; j++) { minv[j] = 3.4e38f; mini[j] = 0; }

    // bf16 codebook viewed as uint4: 1024 uint4 per 32-code chunk (32 per row)
    const uint4* cbk4 = reinterpret_cast<const uint4*>(bfcb);

    // Warm WGP$ with chunk 0 (one 64B prefetch per thread covers 16 KB)
    __builtin_prefetch(reinterpret_cast<const char*>(cbk4) + tid * 64, 0, 3);

    for (int cbi = 0; cbi < KCODES / 32; cbi++) {
        __syncthreads();  // previous compute done reading Bbf
        // Synchronous copy global(bf16, prefetched -> WGP$-hot) -> LDS.
        // No long-lived registers: nothing for the allocator to spill.
#pragma unroll
        for (int u = 0; u < 4; u++) {
            const int i  = tid + 256 * u;
            const int r  = i >> 5;         // 32 uint4 per 32-code row
            const int c8 = (i & 31) * 8;   // half offset (16B granules)
            *reinterpret_cast<uint4*>(&Bbf[r][c8]) = cbk4[cbi * 1024 + i];
        }
        __syncthreads();

        // Prefetch next chunk's cachelines; overlaps the WMMA loop below.
        if (cbi + 1 < KCODES / 32) {
            __builtin_prefetch(
                reinterpret_cast<const char*>(cbk4 + (cbi + 1) * 1024) + tid * 64, 0, 3);
        }

        // Per-lane norm for this chunk's column (L2/WGP$-hot, overlaps compute)
        const float nk = norms[cbi * 32 + codeLocal];

        // B fragments double-buffered: loads for step ks+1 issue before WMMA ks.
        Frag bb[2];
        bb[0].q[0] = *reinterpret_cast<const uint4*>(&Bbf[codeLocal][hl * 16]);
        bb[0].q[1] = *reinterpret_cast<const uint4*>(&Bbf[codeLocal][hl * 16 + 8]);

        v8f acc = {0.f, 0.f, 0.f, 0.f, 0.f, 0.f, 0.f, 0.f};
#pragma unroll
        for (int ks = 0; ks < 8; ks++) {
            if (ks + 1 < 8) {
                const int kb = (ks + 1) * 32;
                // B k order used here (contiguous, same as A): lanes 0-15 K=kb..kb+15; lanes 16-31 K=kb+16..kb+31
                bb[(ks + 1) & 1].q[0] =
                    *reinterpret_cast<const uint4*>(&Bbf[codeLocal][kb + hl * 16]);
                bb[(ks + 1) & 1].q[1] =
                    *reinterpret_cast<const uint4*>(&Bbf[codeLocal][kb + hl * 16 + 8]);
            }
            acc = __builtin_amdgcn_wmma_f32_16x16x32_bf16(
                false, aF[ks].v, false, bb[ks & 1].v, (short)0, acc, false, false);
        }

        // acc[j]: row = rt*16 + j + hl*8, code = cbi*32 + ch*16 + m
        const int codeG = cbi * 32 + codeLocal;
#pragma unroll
        for (int j = 0; j < 8; j++) {
            const float s = nk - 2.0f * acc[j];
            if (s < minv[j]) { minv[j] = s; mini[j] = codeG; }  // strict < keeps first index
        }
    }

    // Reduce across the 16 lanes of each half-wave (rows identical, codes differ)
#pragma unroll
    for (int j = 0; j < 8; j++) {
        float v  = minv[j];
        int   ix = mini[j];
        for (int off = 1; off < 16; off <<= 1) {
            const float ov = __shfl_xor(v, off, 32);
            const int   oi = __shfl_xor(ix, off, 32);
            if (ov < v || (ov == v && oi < ix)) { v = ov; ix = oi; }
        }
        minv[j] = v; mini[j] = ix;
    }
    if (m == 0) {
#pragma unroll
        for (int j = 0; j < 8; j++) {
            const int rloc = rt * 16 + hl * 8 + j;
            cval[ch][rloc] = minv[j];
            cind[ch][rloc] = mini[j];
        }
    }
    __syncthreads();
    if (tid < MBLK) {
        const float v0 = cval[0][tid]; const int i0 = cind[0][tid];
        const float v1 = cval[1][tid]; const int i1 = cind[1][tid];
        finalIdx[tid] = (v1 < v0 || (v1 == v0 && i1 < i0)) ? i1 : i0;
    }
    __syncthreads();

    // Fused gather/copy/loss phase: z rows are L2/WGP$-hot from staging above.
    // Non-temporal output stores: don't evict the L2-resident codebook / z lines.
    float lsum = 0.f;
    for (int rr = 0; rr < MBLK; rr++) {
        const size_t row = (size_t)(rowBase + rr);
        const int k = finalIdx[rr];
        const float zv = z[row * DIM + tid];
        const float qv = cbk[(size_t)k * DIM + tid];
        __builtin_nontemporal_store(zv, &outZ[row * DIM + tid]);
        __builtin_nontemporal_store(qv, &outQ[row * DIM + tid]);  // z_q_st == z_q numerically
        const float d = qv - zv;
        lsum += d * d;
    }
    red[tid] = lsum;
    __syncthreads();
    for (int s = 128; s > 0; s >>= 1) {
        if (tid < s) red[tid] += red[tid + s];
        __syncthreads();
    }
    if (tid == 0) partial[blockIdx.x] = red[0];
}

// ---------------------------------------------------------------------------
// Kernel 2: deterministic final reduction; loss = 2 * mean((z_q - z_e)^2)
// ---------------------------------------------------------------------------
__global__ __launch_bounds__(256) void loss_kernel(const float* __restrict__ partial,
                                                   float* __restrict__ lossOut) {
    __shared__ float red[256];
    const int tid = threadIdx.x;
    float s = partial[tid] + partial[tid + 256];
    red[tid] = s;
    __syncthreads();
    for (int st = 128; st > 0; st >>= 1) {
        if (tid < st) red[tid] += red[tid + st];
        __syncthreads();
    }
    if (tid == 0) lossOut[0] = 2.0f * red[0] / (float)((size_t)N_ROWS * DIM);
}

extern "C" void kernel_launch(void* const* d_in, const int* in_sizes, int n_in,
                              void* d_out, int out_size, void* d_ws, size_t ws_size,
                              hipStream_t stream) {
    const float* z   = (const float*)d_in[0];   // [32768, 256] f32
    const float* cbk = (const float*)d_in[1];   // [1024, 256] f32

    float* out = (float*)d_out;
    // Output layout: z_e (N*D) | loss (1) | z_q_st (N*D)
    float* outZ    = out;
    float* lossOut = out + (size_t)N_ROWS * DIM;
    float* outQ    = lossOut + 1;

    char*           ws      = (char*)d_ws;
    float*          norms   = (float*)ws;                    // 1024 f32   (4 KB)
    float*          partial = (float*)(ws + 4096);           // 512 f32    (2 KB)
    unsigned short* bfcb    = (unsigned short*)(ws + 8192);  // 1024x256 bf16 (512 KB)

    cbprep_kernel<<<KCODES, 256, 0, stream>>>(cbk, norms, bfcb);
    vq_fused_kernel<<<N_ROWS / MBLK, 256, 0, stream>>>(z, cbk, norms, bfcb,
                                                       outZ, outQ, partial);
    loss_kernel<<<1, 256, 0, stream>>>(partial, lossOut);
}
